// DPR_79654463471807
// MI455X (gfx1250) — matrix operand order repaired and checked by reading, not compiled, not run on hardware
//
#include <hip/hip_runtime.h>
#include <math.h>

#define B_    8
#define N_    4096
#define C_    768
#define RD_   192
#define M_    64
#define SCALE_ 0.07216878364870322f   // 192^-0.5

typedef __attribute__((ext_vector_type(16))) _Float16 v16h;
typedef __attribute__((ext_vector_type(8)))  _Float16 v8h;
typedef __attribute__((ext_vector_type(8)))  float    v8f;

// ---------------------------------------------------------------------------
// Unified f16 WMMA GEMM:  D[M,N] (f32 acc) = A[M,K] * Bt[N,K]^T
//   256 threads = 8 waves in a 4 (row) x 2 (col) grid; each wave owns a
//   32x32 output tile (2x2 register tiling -> 4 WMMAs per 8 b128 loads),
//   block tile = 128 rows x 64 cols.
//   modes: 0 = store f32 row-major
//          1 = store f16 row-major
//          2 = store f16 transposed per-batch: Oh[(batch*tPC + n)*tPN + rowIn]
//          3 = bias + exact GELU -> f16 row-major
// ---------------------------------------------------------------------------
__device__ __forceinline__ void store_tile(v8f acc, int mode,
                                           float* __restrict__ Of,
                                           _Float16* __restrict__ Oh,
                                           const float* __restrict__ bias,
                                           long ldc, int rowg, int rowIn, int n,
                                           int batch, long tPC, long tPN, int kh) {
  if (mode == 2) {
    _Float16 tmp[8];
#pragma unroll
    for (int r = 0; r < 8; ++r) tmp[r] = (_Float16)acc[r];
    v8h* dst = (v8h*)(Oh + ((long)batch * tPC + n) * tPN + rowIn + kh * 8);
    *dst = *(const v8h*)tmp;
  } else if (mode == 0) {
#pragma unroll
    for (int r = 0; r < 8; ++r)
      Of[(long)(rowg + kh * 8 + r) * ldc + n] = acc[r];
  } else if (mode == 1) {
#pragma unroll
    for (int r = 0; r < 8; ++r)
      Oh[(long)(rowg + kh * 8 + r) * ldc + n] = (_Float16)acc[r];
  } else {  // bias + exact GELU
    const float bb = bias[n];
#pragma unroll
    for (int r = 0; r < 8; ++r) {
      float t = acc[r] + bb;
      float g = 0.5f * t * (1.f + erff(t * 0.70710678118654752f));
      Oh[(long)(rowg + kh * 8 + r) * ldc + n] = (_Float16)g;
    }
  }
}

__global__ void gemm_f16(const _Float16* __restrict__ A, int lda,
                         const _Float16* __restrict__ Bt, int ldb,
                         int K, int ldc,
                         float* __restrict__ Of, _Float16* __restrict__ Oh,
                         const float* __restrict__ bias,
                         int batch_rows, long aStride, long bStride,
                         int mode, long tPC, long tPN) {
  const int lane = threadIdx.x & 31;
  const int wv   = threadIdx.x >> 5;            // 8 waves: 4 row-groups x 2 col-groups
  const int row0 = blockIdx.y * 128 + (wv & 3) * 32;
  const int col0 = blockIdx.x * 64  + (wv >> 2) * 32;
  int batch = 0, rowIn = row0;
  if (batch_rows > 0) { batch = row0 / batch_rows; rowIn = row0 - batch * batch_rows; }
  const int ml = lane & 15;
  const int kh = lane >> 4;
  const _Float16* ap0 = A  + (long)batch * aStride + (long)(rowIn + ml) * lda + kh * 8;
  const _Float16* ap1 = ap0 + (long)16 * lda;
  const _Float16* bp0 = Bt + (long)batch * bStride + (long)(col0 + ml) * ldb + kh * 8;
  const _Float16* bp1 = bp0 + (long)16 * ldb;
  v8f acc00 = {0.f,0.f,0.f,0.f,0.f,0.f,0.f,0.f};
  v8f acc01 = acc00, acc10 = acc00, acc11 = acc00;
  for (int k0 = 0; k0 < K; k0 += 32) {
    v8h a0l = *(const v8h*)ap0;          // k = k0 + kh*8 .. +7
    v8h a0h = *(const v8h*)(ap0 + 16);   // k = k0 + 16 + kh*8 .. +7
    v8h a1l = *(const v8h*)ap1;
    v8h a1h = *(const v8h*)(ap1 + 16);
    v8h b0l = *(const v8h*)bp0;          // k = k0 + kh*8 .. +7
    v8h b0h = *(const v8h*)(bp0 + 16);   // k = k0 + 16 + kh*8 .. +7
    v8h b1l = *(const v8h*)bp1;
    v8h b1h = *(const v8h*)(bp1 + 16);
    v16h a0 = __builtin_shufflevector(a0l, a0h, 0,1,2,3,4,5,6,7,8,9,10,11,12,13,14,15);
    v16h a1 = __builtin_shufflevector(a1l, a1h, 0,1,2,3,4,5,6,7,8,9,10,11,12,13,14,15);
    v16h b0 = __builtin_shufflevector(b0l, b0h, 0,1,2,3,4,5,6,7,8,9,10,11,12,13,14,15);
    v16h b1 = __builtin_shufflevector(b1l, b1h, 0,1,2,3,4,5,6,7,8,9,10,11,12,13,14,15);
    acc00 = __builtin_amdgcn_wmma_f32_16x16x32_f16(false, a0, false, b0, (short)0, acc00, false, false);
    acc01 = __builtin_amdgcn_wmma_f32_16x16x32_f16(false, a0, false, b1, (short)0, acc01, false, false);
    acc10 = __builtin_amdgcn_wmma_f32_16x16x32_f16(false, a1, false, b0, (short)0, acc10, false, false);
    acc11 = __builtin_amdgcn_wmma_f32_16x16x32_f16(false, a1, false, b1, (short)0, acc11, false, false);
    ap0 += 32; ap1 += 32; bp0 += 32; bp1 += 32;
  }
  const int n0 = col0 + ml, n1 = col0 + 16 + ml;
  store_tile(acc00, mode, Of, Oh, bias, ldc, row0,      rowIn,      n0, batch, tPC, tPN, kh);
  store_tile(acc01, mode, Of, Oh, bias, ldc, row0,      rowIn,      n1, batch, tPC, tPN, kh);
  store_tile(acc10, mode, Of, Oh, bias, ldc, row0 + 16, rowIn + 16, n0, batch, tPC, tPN, kh);
  store_tile(acc11, mode, Of, Oh, bias, ldc, row0 + 16, rowIn + 16, n1, batch, tPC, tPN, kh);
}

// LayerNorm over C=768 per token; emits f16 copies of x and LN(x).
__global__ void ln_prep(const float* __restrict__ x, const float* __restrict__ g,
                        const float* __restrict__ b,
                        _Float16* __restrict__ x_h, _Float16* __restrict__ xln_h) {
  const long row = blockIdx.x;
  const int  t   = threadIdx.x;            // 256
  const float* xr = x + row * C_;
  const float v0 = xr[t], v1 = xr[t + 256], v2 = xr[t + 512];
  __shared__ float s1[256], s2[256];
  s1[t] = v0 + v1 + v2;
  s2[t] = v0 * v0 + v1 * v1 + v2 * v2;
  __syncthreads();
  for (int o = 128; o > 0; o >>= 1) {
    if (t < o) { s1[t] += s1[t + o]; s2[t] += s2[t + o]; }
    __syncthreads();
  }
  const float mu  = s1[0] * (1.f / C_);
  const float var = s2[0] * (1.f / C_) - mu * mu;
  const float r   = rsqrtf(var + 1e-5f);
  _Float16* xh = x_h   + row * C_;
  _Float16* lh = xln_h + row * C_;
  xh[t] = (_Float16)v0; xh[t + 256] = (_Float16)v1; xh[t + 512] = (_Float16)v2;
  lh[t]       = (_Float16)((v0 - mu) * r * g[t]       + b[t]);
  lh[t + 256] = (_Float16)((v1 - mu) * r * g[t + 256] + b[t + 256]);
  lh[t + 512] = (_Float16)((v2 - mu) * r * g[t + 512] + b[t + 512]);
}

// Tiled LDS transpose: x[b,n,c] (f32) -> xT[b,c,n] (f16)
__global__ void transpose_x(const float* __restrict__ x, _Float16* __restrict__ xT) {
  __shared__ _Float16 tile[32][33];
  const int b  = blockIdx.z;
  const int n0 = blockIdx.x * 32, c0 = blockIdx.y * 32;
  for (int i = threadIdx.y; i < 32; i += 8)
    tile[i][threadIdx.x] =
        (_Float16)x[((long)(b * N_ + n0 + i)) * C_ + c0 + threadIdx.x];
  __syncthreads();
  for (int i = threadIdx.y; i < 32; i += 8)
    xT[((long)(b * C_ + c0 + i)) * N_ + n0 + threadIdx.x] = tile[threadIdx.x][i];
}

// w[K,N] f32 -> wt[N,K] f16
__global__ void wt_kernel(const float* __restrict__ w, _Float16* __restrict__ wt,
                          int K, int N) {
  const long i = (long)blockIdx.x * 256 + threadIdx.x;
  if (i >= (long)K * N) return;
  const long n = i / K, k = i - n * K;
  wt[i] = (_Float16)w[k * N + n];
}

// softmax over M=64 per token; writes assignment^T [b,m,n] as f16
__global__ void softmax_assign(const float* __restrict__ logits,
                               _Float16* __restrict__ assignT) {
  const long row = blockIdx.x;          // global token index
  const int  m   = threadIdx.x;         // 64
  const float l  = logits[row * M_ + m];
  __shared__ float sh[64];
  sh[m] = l; __syncthreads();
  for (int o = 32; o > 0; o >>= 1) { if (m < o) sh[m] = fmaxf(sh[m], sh[m + o]); __syncthreads(); }
  const float mx = sh[0]; __syncthreads();
  const float e = expf(l - mx);
  sh[m] = e; __syncthreads();
  for (int o = 32; o > 0; o >>= 1) { if (m < o) sh[m] += sh[m + o]; __syncthreads(); }
  const float p = e / sh[0];
  const long b = row >> 12, n = row & (N_ - 1);
  assignT[((b * M_ + m) << 12) + n] = (_Float16)p;
}

__global__ void proto_weight(const _Float16* __restrict__ assignT,
                             float* __restrict__ wgt) {
  const long row = blockIdx.x;          // b*M + m
  const int  t   = threadIdx.x;         // 256
  float s = 0.f;
  for (int i = t; i < N_; i += 256) s += (float)assignT[row * N_ + i];
  __shared__ float sh[256];
  sh[t] = s; __syncthreads();
  for (int o = 128; o > 0; o >>= 1) { if (t < o) sh[t] += sh[t + o]; __syncthreads(); }
  if (t == 0) wgt[row] = fmaxf(sh[0], 1e-6f);
}

// prototypes0 = l2norm(LN(content/weight)); seed = prototypes0 + proto_q
__global__ void proto_norm1(const float* __restrict__ content, const float* __restrict__ wgt,
                            const float* __restrict__ png, const float* __restrict__ pnb,
                            const float* __restrict__ proto_q,
                            float* __restrict__ proto0, _Float16* __restrict__ seed_h) {
  const int row = blockIdx.x;           // b*M + m
  const int t   = threadIdx.x;          // 256
  const float iw = 1.f / wgt[row];
  const float a0 = content[row * C_ + t]       * iw;
  const float a1 = content[row * C_ + t + 256] * iw;
  const float a2 = content[row * C_ + t + 512] * iw;
  __shared__ float s1[256], s2[256];
  s1[t] = a0 + a1 + a2;
  s2[t] = a0 * a0 + a1 * a1 + a2 * a2;
  __syncthreads();
  for (int o = 128; o > 0; o >>= 1) {
    if (t < o) { s1[t] += s1[t + o]; s2[t] += s2[t + o]; }
    __syncthreads();
  }
  const float mu  = s1[0] * (1.f / C_);
  const float var = s2[0] * (1.f / C_) - mu * mu;
  const float r   = rsqrtf(var + 1e-5f);
  const float y0 = (a0 - mu) * r * png[t]       + pnb[t];
  const float y1 = (a1 - mu) * r * png[t + 256] + pnb[t + 256];
  const float y2 = (a2 - mu) * r * png[t + 512] + pnb[t + 512];
  __syncthreads();
  s1[t] = y0 * y0 + y1 * y1 + y2 * y2;
  __syncthreads();
  for (int o = 128; o > 0; o >>= 1) { if (t < o) s1[t] += s1[t + o]; __syncthreads(); }
  const float rn = 1.f / fmaxf(sqrtf(s1[0]), 1e-12f);
  const int m = row & (M_ - 1);
  proto0[row * C_ + t]       = y0 * rn;
  proto0[row * C_ + t + 256] = y1 * rn;
  proto0[row * C_ + t + 512] = y2 * rn;
  seed_h[row * C_ + t]       = (_Float16)(y0 * rn + proto_q[m * C_ + t]);
  seed_h[row * C_ + t + 256] = (_Float16)(y1 * rn + proto_q[m * C_ + t + 256]);
  seed_h[row * C_ + t + 512] = (_Float16)(y2 * rn + proto_q[m * C_ + t + 512]);
}

// softmax over N=4096 with SCALE, f16 output (A operand of attn@v)
__global__ void softmax_attn(const float* __restrict__ logits,
                             _Float16* __restrict__ attn_h) {
  const long row = blockIdx.x;          // 512 rows
  const int  t   = threadIdx.x;         // 256
  float l[16];
  float mx = -1e30f;
#pragma unroll
  for (int i = 0; i < 16; ++i) {
    l[i] = logits[row * N_ + t + i * 256] * SCALE_;
    mx = fmaxf(mx, l[i]);
  }
  __shared__ float sh[256];
  sh[t] = mx; __syncthreads();
  for (int o = 128; o > 0; o >>= 1) { if (t < o) sh[t] = fmaxf(sh[t], sh[t + o]); __syncthreads(); }
  mx = sh[0]; __syncthreads();
  float s = 0.f;
#pragma unroll
  for (int i = 0; i < 16; ++i) { l[i] = expf(l[i] - mx); s += l[i]; }
  sh[t] = s; __syncthreads();
  for (int o = 128; o > 0; o >>= 1) { if (t < o) sh[t] += sh[t + o]; __syncthreads(); }
  const float inv = 1.f / sh[0];
#pragma unroll
  for (int i = 0; i < 16; ++i)
    attn_h[row * N_ + t + i * 256] = (_Float16)(l[i] * inv);
}

// prototypes = l2norm(LN(proto0 + sigmoid(gate)*refine)); writes final f32 + f16
__global__ void proto_norm2(const float* __restrict__ proto0, const float* __restrict__ refine,
                            const float* __restrict__ png, const float* __restrict__ pnb,
                            const float* __restrict__ gate,
                            float* __restrict__ outProto, _Float16* __restrict__ proto2_h) {
  const int row = blockIdx.x;
  const int t   = threadIdx.x;          // 256
  const float gamma = 1.f / (1.f + expf(-gate[0]));
  const float a0 = proto0[row * C_ + t]       + gamma * refine[row * C_ + t];
  const float a1 = proto0[row * C_ + t + 256] + gamma * refine[row * C_ + t + 256];
  const float a2 = proto0[row * C_ + t + 512] + gamma * refine[row * C_ + t + 512];
  __shared__ float s1[256], s2[256];
  s1[t] = a0 + a1 + a2;
  s2[t] = a0 * a0 + a1 * a1 + a2 * a2;
  __syncthreads();
  for (int o = 128; o > 0; o >>= 1) {
    if (t < o) { s1[t] += s1[t + o]; s2[t] += s2[t + o]; }
    __syncthreads();
  }
  const float mu  = s1[0] * (1.f / C_);
  const float var = s2[0] * (1.f / C_) - mu * mu;
  const float r   = rsqrtf(var + 1e-5f);
  const float y0 = (a0 - mu) * r * png[t]       + pnb[t];
  const float y1 = (a1 - mu) * r * png[t + 256] + pnb[t + 256];
  const float y2 = (a2 - mu) * r * png[t + 512] + pnb[t + 512];
  __syncthreads();
  s1[t] = y0 * y0 + y1 * y1 + y2 * y2;
  __syncthreads();
  for (int o = 128; o > 0; o >>= 1) { if (t < o) s1[t] += s1[t + o]; __syncthreads(); }
  const float rn = 1.f / fmaxf(sqrtf(s1[0]), 1e-12f);
  outProto[row * C_ + t]       = y0 * rn;
  outProto[row * C_ + t + 256] = y1 * rn;
  outProto[row * C_ + t + 512] = y2 * rn;
  proto2_h[row * C_ + t]       = (_Float16)(y0 * rn);
  proto2_h[row * C_ + t + 256] = (_Float16)(y1 * rn);
  proto2_h[row * C_ + t + 512] = (_Float16)(y2 * rn);
}

// l2-normalize rows of length 192 (f32 -> f16)
__global__ void l2rows(const float* __restrict__ in, _Float16* __restrict__ out) {
  const long row = blockIdx.x;
  const int  t   = threadIdx.x;         // 64
  const float a0 = in[row * RD_ + t], a1 = in[row * RD_ + t + 64], a2 = in[row * RD_ + t + 128];
  __shared__ float sh[64];
  sh[t] = a0 * a0 + a1 * a1 + a2 * a2; __syncthreads();
  for (int o = 32; o > 0; o >>= 1) { if (t < o) sh[t] += sh[t + o]; __syncthreads(); }
  const float rn = 1.f / fmaxf(sqrtf(sh[0]), 1e-12f);
  out[row * RD_ + t]       = (_Float16)(a0 * rn);
  out[row * RD_ + t + 64]  = (_Float16)(a1 * rn);
  out[row * RD_ + t + 128] = (_Float16)(a2 * rn);
}

// softmax over 64, max-prob, argmax (first-index ties), sort key
__global__ void score_post(const float* __restrict__ logits,
                           float* __restrict__ xs, int* __restrict__ belong,
                           float* __restrict__ key) {
  const long row = blockIdx.x;
  const int  m   = threadIdx.x;         // 64
  const float l  = logits[row * M_ + m] * SCALE_;
  __shared__ float sv[64];
  __shared__ int   si[64];
  sv[m] = l; __syncthreads();
  for (int o = 32; o > 0; o >>= 1) { if (m < o) sv[m] = fmaxf(sv[m], sv[m + o]); __syncthreads(); }
  const float mx = sv[0]; __syncthreads();
  sv[m] = expf(l - mx); __syncthreads();
  for (int o = 32; o > 0; o >>= 1) { if (m < o) sv[m] += sv[m + o]; __syncthreads(); }
  const float s = sv[0]; __syncthreads();
  sv[m] = l; si[m] = m; __syncthreads();
  for (int o = 32; o > 0; o >>= 1) {
    if (m < o) {
      const float ov = sv[m + o]; const int oi = si[m + o];
      if (ov > sv[m] || (ov == sv[m] && oi < si[m])) { sv[m] = ov; si[m] = oi; }
    }
    __syncthreads();
  }
  if (m == 0) {
    const float p = 1.f / s;            // exp(mx-mx)/sum
    xs[row]     = p;
    belong[row] = si[0];
    key[row]    = (float)si[0] + 0.5f * (1.f - p);
  }
}

// per-batch bitonic sort of 4096 (key, idx) pairs in LDS
__global__ void sort_kernel(const float* __restrict__ key, int* __restrict__ sortedIdx) {
  const int b = blockIdx.x;
  const int t = threadIdx.x;            // 1024
  __shared__ float sk[N_];
  __shared__ int   si[N_];
  for (int i = t; i < N_; i += 1024) { sk[i] = key[b * N_ + i]; si[i] = i; }
  __syncthreads();
  for (int k = 2; k <= N_; k <<= 1)
    for (int j = k >> 1; j > 0; j >>= 1) {
      for (int i = t; i < N_; i += 1024) {
        const int ixj = i ^ j;
        if (ixj > i) {
          const bool up = ((i & k) == 0);
          const float a = sk[i], c = sk[ixj];
          if ((a > c) == up) {
            sk[i] = c; sk[ixj] = a;
            const int tmp = si[i]; si[i] = si[ixj]; si[ixj] = tmp;
          }
        }
      }
      __syncthreads();
    }
  for (int i = t; i < N_; i += 1024) sortedIdx[b * N_ + i] = si[i];
}

// gather sorted outputs into d_out
__global__ void gather_out(const float* __restrict__ x, const int* __restrict__ sortedIdx,
                           const int* __restrict__ belong, const float* __restrict__ xs,
                           float* __restrict__ out) {
  const long g = blockIdx.x;            // b*N + position
  const int  t = threadIdx.x;           // 256
  const long b = g >> 12;
  const int  j = sortedIdx[g];
  const float* src = x + ((b << 12) + j) * (long)C_;
  float* dst = out + g * (long)C_;
  for (int c = t; c < C_; c += 256) dst[c] = src[c];
  if (t == 0) {
    float* o1 = out + (long)B_ * N_ * C_;
    float* o2 = o1 + (long)B_ * N_;
    float* o3 = o2 + (long)B_ * N_;
    o1[g] = (float)j;
    o2[g] = (float)belong[(b << 12) + j];
    o3[g] = xs[(b << 12) + j];
  }
}

// ---------------------------------------------------------------------------
extern "C" void kernel_launch(void* const* d_in, const int* in_sizes, int n_in,
                              void* d_out, int out_size, void* d_ws, size_t ws_size,
                              hipStream_t stream) {
  (void)in_sizes; (void)n_in; (void)out_size; (void)ws_size;
  const float* x        = (const float*)d_in[0];
  const float* ln1_g    = (const float*)d_in[1];
  const float* ln1_b    = (const float*)d_in[2];
  const float* w_embed  = (const float*)d_in[3];
  const float* b_embed  = (const float*)d_in[4];
  const float* w_assign = (const float*)d_in[5];
  const float* proto_q  = (const float*)d_in[6];
  const float* w_q      = (const float*)d_in[7];
  const float* w_k      = (const float*)d_in[8];
  const float* w_v      = (const float*)d_in[9];
  const float* w_tok    = (const float*)d_in[10];
  const float* w_proto  = (const float*)d_in[11];
  const float* pn_g     = (const float*)d_in[12];
  const float* pn_b     = (const float*)d_in[13];
  const float* gate     = (const float*)d_in[14];
  float* out = (float*)d_out;

  char* ws = (char*)d_ws;
  size_t off = 0;
  auto alloc = [&](size_t n) {
    off = (off + 255) & ~(size_t)255;
    size_t o = off; off += n; return o;
  };
  const long T = (long)B_ * N_;   // 32768 tokens
  const long P = (long)B_ * M_;   // 512 prototypes

  _Float16* x_h    = (_Float16*)(ws + alloc(T * C_ * 2));
  _Float16* xln_h  = (_Float16*)(ws + alloc(T * C_ * 2));
  _Float16* xT_h   = (_Float16*)(ws + alloc(T * C_ * 2));
  _Float16* emb_h  = (_Float16*)(ws + alloc(T * C_ * 2));
  _Float16* k_h    = (_Float16*)(ws + alloc(T * RD_ * 2));
  _Float16* tok_h  = (_Float16*)(ws + alloc(T * RD_ * 2));
  _Float16* asT_h  = (_Float16*)(ws + alloc(P * N_ * 2));
  _Float16* attn_h = (_Float16*)(ws + alloc(P * N_ * 2));
  _Float16* seed_h = (_Float16*)(ws + alloc(P * C_ * 2));
  _Float16* pro2_h = (_Float16*)(ws + alloc(P * C_ * 2));
  _Float16* q_h    = (_Float16*)(ws + alloc(P * RD_ * 2));
  _Float16* pf_h   = (_Float16*)(ws + alloc(P * RD_ * 2));
  _Float16* wembT  = (_Float16*)(ws + alloc((long)C_ * C_ * 2));
  _Float16* wasgT  = (_Float16*)(ws + alloc((long)C_ * M_ * 2));
  _Float16* wqT    = (_Float16*)(ws + alloc((long)C_ * RD_ * 2));
  _Float16* wkT    = (_Float16*)(ws + alloc((long)C_ * RD_ * 2));
  _Float16* wvT    = (_Float16*)(ws + alloc((long)C_ * C_ * 2));
  _Float16* wtokT  = (_Float16*)(ws + alloc((long)C_ * RD_ * 2));
  _Float16* wproT  = (_Float16*)(ws + alloc((long)C_ * RD_ * 2));
  float* logitsA = (float*)(ws + alloc(T * M_ * 4));
  float* attnL   = (float*)(ws + alloc(P * N_ * 4));
  float* content = (float*)(ws + alloc(P * C_ * 4));
  float* refine  = (float*)(ws + alloc(P * C_ * 4));
  float* proto0  = (float*)(ws + alloc(P * C_ * 4));
  float* tokF    = (float*)(ws + alloc(T * RD_ * 4));
  float* pfF     = (float*)(ws + alloc(P * RD_ * 4));
  float* wgt     = (float*)(ws + alloc(P * 4));
  float* xs      = (float*)(ws + alloc(T * 4));
  int*   belong  = (int*)(ws + alloc(T * 4));
  float* skey    = (float*)(ws + alloc(T * 4));
  int*   sidx    = (int*)(ws + alloc(T * 4));
  // lifetime aliases (xln dead after embed GEMM -> reuse for v^T; logitsA reused for scores)
  _Float16* vT_h    = xln_h;
  float*    scoresL = logitsA;
  float*    outProto = out + (long)B_ * N_ * C_ + 3L * B_ * N_;

  auto gemm = [&](const _Float16* A, int lda, const _Float16* Bt, int ldb,
                  long Mtot, int Nn, int K, float* Of, _Float16* Oh, const float* bias,
                  int batch_rows, long aStride, long bStride, int mode, long tPC, long tPN) {
    dim3 grid(Nn / 64, (unsigned)(Mtot / 128));
    gemm_f16<<<grid, 256, 0, stream>>>(A, lda, Bt, ldb, K, Nn, Of, Oh, bias,
                                       batch_rows, aStride, bStride, mode, tPC, tPN);
  };
  auto wt = [&](const float* w, _Float16* wtp, int K, int Nn) {
    long total = (long)K * Nn;
    wt_kernel<<<(unsigned)((total + 255) / 256), 256, 0, stream>>>(w, wtp, K, Nn);
  };

  // --- weight transposes (f32 [K,N] -> f16 [N,K]) ---
  wt(w_embed, wembT, C_, C_);   wt(w_assign, wasgT, C_, M_);
  wt(w_q, wqT, C_, RD_);        wt(w_k, wkT, C_, RD_);
  wt(w_v, wvT, C_, C_);         wt(w_tok, wtokT, C_, RD_);
  wt(w_proto, wproT, C_, RD_);

  // --- LN + f16 conversion, x transpose ---
  ln_prep<<<(unsigned)T, 256, 0, stream>>>(x, ln1_g, ln1_b, x_h, xln_h);
  transpose_x<<<dim3(N_ / 32, C_ / 32, B_), dim3(32, 8), 0, stream>>>(x, xT_h);

  // embed = GELU(LN(x) @ w_embed + b)
  gemm(xln_h, C_, wembT, C_, T, C_, C_, nullptr, emb_h, b_embed, 0, 0, 0, 3, 0, 0);
  // assignment logits = embed @ w_assign
  gemm(emb_h, C_, wasgT, C_, T, M_, C_, logitsA, nullptr, nullptr, 0, 0, 0, 0, 0, 0);
  softmax_assign<<<(unsigned)T, 64, 0, stream>>>(logitsA, asT_h);
  proto_weight<<<(unsigned)P, 256, 0, stream>>>(asT_h, wgt);
  // proto_content = assignment^T @ x   (per batch)
  gemm(asT_h, N_, xT_h, N_, P, C_, N_, content, nullptr, nullptr,
       M_, (long)M_ * N_, (long)C_ * N_, 0, 0, 0);
  proto_norm1<<<(unsigned)P, 256, 0, stream>>>(content, wgt, pn_g, pn_b, proto_q,
                                               proto0, seed_h);
  // q / k / v projections
  gemm(seed_h, C_, wqT, C_, P, RD_, C_, nullptr, q_h, nullptr, 0, 0, 0, 1, 0, 0);
  gemm(emb_h, C_, wkT, C_, T, RD_, C_, nullptr, k_h, nullptr, 0, 0, 0, 1, 0, 0);
  gemm(x_h, C_, wvT, C_, T, C_, C_, nullptr, vT_h, nullptr,
       N_, (long)N_ * C_, 0, 2, C_, N_);          // transposed store -> v^T [b,c,n]
  // attention logits = q @ k^T (per batch), softmax, proto_refine = attn @ v
  gemm(q_h, RD_, k_h, RD_, P, N_, RD_, attnL, nullptr, nullptr,
       M_, (long)M_ * RD_, (long)N_ * RD_, 0, 0, 0);
  softmax_attn<<<(unsigned)P, 256, 0, stream>>>(attnL, attn_h);
  gemm(attn_h, N_, vT_h, N_, P, C_, N_, refine, nullptr, nullptr,
       M_, (long)M_ * N_, (long)C_ * N_, 0, 0, 0);
  proto_norm2<<<(unsigned)P, 256, 0, stream>>>(proto0, refine, pn_g, pn_b, gate,
                                               outProto, pro2_h);
  // token / prototype features + scores
  gemm(emb_h, C_, wtokT, C_, T, RD_, C_, tokF, nullptr, nullptr, 0, 0, 0, 0, 0, 0);
  l2rows<<<(unsigned)T, 64, 0, stream>>>(tokF, tok_h);
  gemm(pro2_h, C_, wproT, C_, P, RD_, C_, pfF, nullptr, nullptr, 0, 0, 0, 0, 0, 0);
  l2rows<<<(unsigned)P, 64, 0, stream>>>(pfF, pf_h);
  gemm(tok_h, RD_, pf_h, RD_, T, M_, RD_, scoresL, nullptr, nullptr,
       N_, (long)N_ * RD_, (long)M_ * RD_, 0, 0, 0);
  score_post<<<(unsigned)T, 64, 0, stream>>>(scoresL, xs, belong, skey);
  // dispatch: per-batch sort + gather
  sort_kernel<<<B_, 1024, 0, stream>>>(skey, sidx);
  gather_out<<<(unsigned)T, 256, 0, stream>>>(x, sidx, belong, xs, out);
}
